// EDENNetwork_26130581029066
// MI455X (gfx1250) — matrix operand order repaired and checked by reading, not compiled, not run on hardware
//
#include <hip/hip_runtime.h>

// ---------------------------------------------------------------------------
// MI455X (gfx1250, wave32) implementation of the EDEN forward pass.
// Dense layers (stem, proj, Q/K/V/O, head) run as bf16 WMMA GEMMs
// (v_wmma_f32_16x16x32_bf16) with DOUBLE-BUFFERED async global->LDS weight
// staging (ASYNCcnt-tracked gfx1250 async copies overlap the WMMA compute of
// the previous K-step). Convs + reductions are fused fp32 kernels. The whole
// ~170MB working set fits in the 192MB L2.
// ---------------------------------------------------------------------------

typedef __attribute__((ext_vector_type(16))) __bf16 bf16x16;
typedef __attribute__((ext_vector_type(8)))  float  floatx8;

#if __has_builtin(__builtin_amdgcn_global_load_async_to_lds_b128) && \
    __has_builtin(__builtin_amdgcn_s_wait_asynccnt)
#define HAVE_ASYNC_LDS 1
// Param types confirmed by compiler diagnostics: (int4 AS1*, int4 AS3*, imm, imm)
typedef int v4i __attribute__((vector_size(4 * sizeof(int))));
typedef __attribute__((address_space(1))) v4i glb_v4i_t;
typedef __attribute__((address_space(3))) v4i lds_v4i_t;
#endif

// ---------------- wave32 reduction helpers ----------------
__device__ inline float wave_sum(float v) {
#pragma unroll
  for (int m = 16; m >= 1; m >>= 1) v += __shfl_xor(v, m, 32);
  return v;
}
__device__ inline float wave_max(float v) {
#pragma unroll
  for (int m = 16; m >= 1; m >>= 1) v = fmaxf(v, __shfl_xor(v, m, 32));
  return v;
}
__device__ inline float wave_min(float v) {
#pragma unroll
  for (int m = 16; m >= 1; m >>= 1) v = fminf(v, __shfl_xor(v, m, 32));
  return v;
}

__device__ inline bf16x16 pack_bf16(float4 a, float4 b, float4 c, float4 d) {
  bf16x16 o;
  o[0]  = (__bf16)a.x; o[1]  = (__bf16)a.y; o[2]  = (__bf16)a.z; o[3]  = (__bf16)a.w;
  o[4]  = (__bf16)b.x; o[5]  = (__bf16)b.y; o[6]  = (__bf16)b.z; o[7]  = (__bf16)b.w;
  o[8]  = (__bf16)c.x; o[9]  = (__bf16)c.y; o[10] = (__bf16)c.z; o[11] = (__bf16)c.w;
  o[12] = (__bf16)d.x; o[13] = (__bf16)d.y; o[14] = (__bf16)d.z; o[15] = (__bf16)d.w;
  return o;
}

// ---------------------------------------------------------------------------
// GEMM:  C[M,N] = act( A[M,K] * W[N,K]^T + bias[N] )
// Block = 256 threads (8 waves). Block tile = 128 (M) x 64 (N).
// Wave w owns M rows [m0+16w, m0+16w+16); 4 accumulators per wave span the
// 64-wide N tile (A fragment reused 4x per K-step).
// Weight slabs (64x32 fp32, 8KB) are staged global->LDS with DOUBLE buffering:
// the async copy for K-step k+1 is issued before computing K-step k, and
// s_wait_asynccnt 0 + s_barrier land only after the 4 WMMAs — the DMA runs
// under the matrix math. Indices are CLAMPED (never predicated) so the K-loop
// is branch-free; masking happens only at the store. K % 32 == 0 required.
// ---------------------------------------------------------------------------
#define GEMM_LDS_STRIDE 36   // floats; 144B rows: 16B-aligned, bank-staggered

__global__ __launch_bounds__(256)
void gemm_wmma_nt(const float* __restrict__ A, const float* __restrict__ W,
                  const float* __restrict__ bias, float* __restrict__ C,
                  int M, int N, int K, int relu)
{
  __shared__ float Bsf[2][64 * GEMM_LDS_STRIDE];

  const int tid  = threadIdx.x;
  const int wv   = tid >> 5;
  const int lane = tid & 31;
  const int half = lane >> 4;       // 0: lanes 0-15, 1: lanes 16-31
  const int r    = lane & 15;
  const int n0   = blockIdx.x * 64;
  const int m0   = blockIdx.y * 128 + wv * 16;

  // A stream: one row per lane, clamped (M is always a multiple of 16 here).
  const int arow = min(m0 + r, M - 1);
  const float* __restrict__ Ap = A + (size_t)arow * K;

  // Cooperative weight staging map: thread -> (row 0..63, col group 0..3)
  const int srow = tid >> 2;
  const int scol = (tid & 3) * 8;
  const int wrow = min(n0 + srow, N - 1);           // clamp ragged N (head)
  const float* __restrict__ Wp = W + (size_t)wrow * K + scol;
  const int sidx = srow * GEMM_LDS_STRIDE + scol;

  floatx8 acc[4];
#pragma unroll
  for (int nt = 0; nt < 4; ++nt)
    acc[nt] = (floatx8){0.f, 0.f, 0.f, 0.f, 0.f, 0.f, 0.f, 0.f};

  auto stage = [&](int buf, int kb) {
#ifdef HAVE_ASYNC_LDS
    __builtin_amdgcn_global_load_async_to_lds_b128(
        (glb_v4i_t*)(Wp + kb),     (lds_v4i_t*)&Bsf[buf][sidx],     0, 0);
    __builtin_amdgcn_global_load_async_to_lds_b128(
        (glb_v4i_t*)(Wp + kb + 4), (lds_v4i_t*)&Bsf[buf][sidx + 4], 0, 0);
#else
    *(float4*)&Bsf[buf][sidx]     = *(const float4*)(Wp + kb);
    *(float4*)&Bsf[buf][sidx + 4] = *(const float4*)(Wp + kb + 4);
#endif
  };

  // Prologue: stage slab 0 and make it visible.
  stage(0, 0);
#ifdef HAVE_ASYNC_LDS
  __builtin_amdgcn_s_wait_asynccnt(0);
#endif
  __syncthreads();

  int p = 0;
  for (int kb = 0; kb < K; kb += 32) {
    const bool more = (kb + 32 < K);
    // Issue async staging of the NEXT slab into the other buffer; it runs
    // concurrently with this K-step's ds_loads + WMMAs.
    if (more) {
      stage(p ^ 1, kb + 32);
      __builtin_prefetch(Ap + kb + 32, 0, 1);
    }

    // A fragment (16x32 bf16): lane holds row arow, K runs
    // [kb+8h, kb+8h+8) and [kb+16+8h, kb+16+8h+8)  (ISA §7.12.2, wave32)
    float4 a0 = *(const float4*)(Ap + kb + half * 8);
    float4 a1 = *(const float4*)(Ap + kb + half * 8 + 4);
    float4 a2 = *(const float4*)(Ap + kb + 16 + half * 8);
    float4 a3 = *(const float4*)(Ap + kb + 16 + half * 8 + 4);
    bf16x16 af = pack_bf16(a0, a1, a2, a3);

#pragma unroll
    for (int nt = 0; nt < 4; ++nt) {
      // B fragment (32x16 bf16): lane holds col (nt*16 + r),
      // K runs [kb+8h, kb+8h+8) and [kb+16+8h, kb+16+8h+8)
      const float* bs = &Bsf[p][(nt * 16 + r) * GEMM_LDS_STRIDE + half * 8];
      float4 b0 = *(const float4*)(bs);
      float4 b1 = *(const float4*)(bs + 4);
      float4 b2 = *(const float4*)(bs + 16);
      float4 b3 = *(const float4*)(bs + 20);
      bf16x16 bfr = pack_bf16(b0, b1, b2, b3);
      acc[nt] = __builtin_amdgcn_wmma_f32_16x16x32_bf16(
          /*neg_a=*/false, af, /*neg_b=*/false, bfr,
          /*c_mod=*/(short)0, acc[nt], /*reuse_a=*/false, /*reuse_b=*/false);
    }

    // Retire this wave's async copies for the next slab, then rendezvous so
    // every wave's portion of Bsf[p^1] is visible (and Bsf[p] is reusable).
#ifdef HAVE_ASYNC_LDS
    if (more) __builtin_amdgcn_s_wait_asynccnt(0);
#endif
    __syncthreads();
    p ^= 1;
  }

  // C/D layout: VGPR i -> (M = m0 + 8*half + i, N = n-tile base + r)
#pragma unroll
  for (int nt = 0; nt < 4; ++nt) {
#pragma unroll
    for (int i = 0; i < 8; ++i) {
      int mm = m0 + half * 8 + i;
      int nn = n0 + nt * 16 + r;
      if (mm < M && nn < N) {
        float v = acc[nt][i] + (bias ? bias[nn] : 0.f);
        if (relu) v = fmaxf(v, 0.f);
        C[(size_t)mm * N + nn] = v;
      }
    }
  }
}

// ---------------------------------------------------------------------------
// Direct 3x3 SAME conv, NCHW / OIHW, fp32. One thread per output element.
// ---------------------------------------------------------------------------
__global__ void conv3x3_same(const float* __restrict__ x, const float* __restrict__ w,
                             const float* __restrict__ bias, float* __restrict__ y,
                             int B, int Cin, int Cout, int H, int Wd)
{
  int idx = blockIdx.x * blockDim.x + threadIdx.x;
  int total = B * Cout * H * Wd;
  if (idx >= total) return;
  int wx = idx % Wd; int t = idx / Wd;
  int hy = t % H;    t /= H;
  int co = t % Cout; int b = t / Cout;

  float s = bias[co];
  for (int ci = 0; ci < Cin; ++ci) {
    const float* __restrict__ xp = x + ((size_t)(b * Cin + ci) * H) * Wd;
    const float* __restrict__ wp = w + ((size_t)(co * Cin + ci) * 9);
#pragma unroll
    for (int kh = 0; kh < 3; ++kh) {
      int iy = hy + kh - 1;
      if (iy < 0 || iy >= H) continue;
#pragma unroll
      for (int kw = 0; kw < 3; ++kw) {
        int ix = wx + kw - 1;
        if (ix < 0 || ix >= Wd) continue;
        s = fmaf(xp[iy * Wd + ix], wp[kh * 3 + kw], s);
      }
    }
  }
  y[idx] = s;
}

// Per-channel batch-norm statistics over (B,H,W). One block per channel.
__global__ void bn_stats(const float* __restrict__ y, float* __restrict__ stats,
                         int B, int C, int HW)
{
  int c = blockIdx.x;
  int n = B * HW;
  float s = 0.f, s2 = 0.f;
  for (int i = threadIdx.x; i < n; i += blockDim.x) {
    int b = i / HW, p = i % HW;
    float v = y[(size_t)(b * C + c) * HW + p];
    s += v; s2 += v * v;
  }
  __shared__ float ls[256], ls2[256];
  ls[threadIdx.x] = s; ls2[threadIdx.x] = s2;
  __syncthreads();
  for (int st = 128; st > 0; st >>= 1) {
    if (threadIdx.x < st) {
      ls[threadIdx.x]  += ls[threadIdx.x + st];
      ls2[threadIdx.x] += ls2[threadIdx.x + st];
    }
    __syncthreads();
  }
  if (threadIdx.x == 0) {
    float mu = ls[0] / n;
    float var = ls2[0] / n - mu * mu;
    stats[c]     = mu;
    stats[C + c] = rsqrtf(var + 1e-5f);
  }
}

// Fused BN + ReLU + 2x2 pool (max or avg). One thread per pooled output.
__global__ void bn_relu_pool(const float* __restrict__ y, const float* __restrict__ stats,
                             const float* __restrict__ g, const float* __restrict__ beta,
                             float* __restrict__ out, int B, int C, int H, int Wd, int avg)
{
  int Ho = H >> 1, Wo = Wd >> 1;
  int idx = blockIdx.x * blockDim.x + threadIdx.x;
  int total = B * C * Ho * Wo;
  if (idx >= total) return;
  int ox = idx % Wo; int t = idx / Wo;
  int oy = t % Ho;   t /= Ho;
  int c  = t % C;    int b = t / C;

  float mu = stats[c], is = stats[C + c], gg = g[c], bb = beta[c];
  const float* __restrict__ src = y + ((size_t)(b * C + c) * H) * Wd;
  float acc = avg ? 0.f : -3.402823e38f;
#pragma unroll
  for (int dy = 0; dy < 2; ++dy)
#pragma unroll
    for (int dx = 0; dx < 2; ++dx) {
      float v = src[(2 * oy + dy) * Wd + (2 * ox + dx)];
      v = fmaxf((v - mu) * is * gg + bb, 0.f);
      acc = avg ? (acc + v) : fmaxf(acc, v);
    }
  out[idx] = avg ? acc * 0.25f : acc;
}

// hvec[b,h] = sum_s gates[s] * stems[b, s*256+h]
__global__ void stems_combine(const float* __restrict__ stems, const float* __restrict__ gates,
                              float* __restrict__ hvec, int B)
{
  int idx = blockIdx.x * blockDim.x + threadIdx.x;
  if (idx >= B * 256) return;
  int h = idx & 255, b = idx >> 8;
  const float* __restrict__ p = stems + (size_t)b * 1024;
  hvec[idx] = gates[0] * p[h] + gates[1] * p[256 + h] +
              gates[2] * p[512 + h] + gates[3] * p[768 + h];
}

// Morphogen: per (b,node) stats over 256 channels, write morph[B,8,8] and
// update h in place: h += 0.1*(morph @ morph_w^T + morph_b).
__global__ __launch_bounds__(256)
void morph_kernel(float* __restrict__ hn, const float* __restrict__ morph_w,
                  const float* __restrict__ morph_b, float* __restrict__ morph_out,
                  int B)
{
  int b    = blockIdx.x;
  int node = threadIdx.x >> 5;
  int lane = threadIdx.x & 31;
  float* __restrict__ row = hn + ((size_t)b * 8 + node) * 256;

  float s = 0.f, s2 = 0.f, sa = 0.f, mx = -3.402823e38f, mn = 3.402823e38f;
#pragma unroll
  for (int j = 0; j < 8; ++j) {
    float v = row[lane + 32 * j];
    s += v; s2 += v * v; sa += fabsf(v);
    mx = fmaxf(mx, v); mn = fminf(mn, v);
  }
  s  = wave_sum(s);  s2 = wave_sum(s2); sa = wave_sum(sa);
  mx = wave_max(mx); mn = wave_min(mn);

  float mean = s * (1.f / 256.f);
  float ms   = s2 * (1.f / 256.f);
  float var  = fmaxf(ms - mean * mean, 0.f);
  float m[8] = { mean, sqrtf(var), mx, mn, sqrtf(ms), sa * (1.f / 256.f), 0.f, 0.f };

  if (lane == 0) {
    float* mo = morph_out + ((size_t)b * 8 + node) * 8;
#pragma unroll
    for (int j = 0; j < 8; ++j) mo[j] = m[j];
  }
#pragma unroll
  for (int j = 0; j < 8; ++j) {
    int c = lane + 32 * j;
    const float* __restrict__ wr = morph_w + c * 8;
    float d = morph_b[c];
#pragma unroll
    for (int k = 0; k < 8; ++k) d = fmaf(m[k], wr[k], d);
    row[c] += 0.1f * d;
  }
}

// Tiny 8-node / 4-head attention core. One block (64 threads) per (b,head).
__global__ __launch_bounds__(64)
void attn_kernel(const float* __restrict__ q, const float* __restrict__ k,
                 const float* __restrict__ v, float* __restrict__ o, int B)
{
  int bh = blockIdx.x;
  int b  = bh >> 2, hH = bh & 3;
  int t  = threadIdx.x;
  __shared__ float att[8][8];
  const size_t base = (size_t)b * 2048 + hH * 64;

  { // scores
    int n = t >> 3, m = t & 7;
    const float* __restrict__ qp = q + base + (size_t)n * 256;
    const float* __restrict__ kp = k + base + (size_t)m * 256;
    float s = 0.f;
#pragma unroll 8
    for (int d = 0; d < 64; ++d) s = fmaf(qp[d], kp[d], s);
    att[n][m] = s * 0.125f;   // 1/sqrt(64)
  }
  __syncthreads();
  if (t < 8) { // row softmax
    float mx = -3.402823e38f;
#pragma unroll
    for (int m = 0; m < 8; ++m) mx = fmaxf(mx, att[t][m]);
    float e[8], se = 0.f;
#pragma unroll
    for (int m = 0; m < 8; ++m) { e[m] = __expf(att[t][m] - mx); se += e[m]; }
    float inv = 1.f / se;
#pragma unroll
    for (int m = 0; m < 8; ++m) att[t][m] = e[m] * inv;
  }
  __syncthreads();
  { // o[n,d] = sum_m att[n][m] * v[m,d]
    int d = t;
    float vv[8];
#pragma unroll
    for (int m = 0; m < 8; ++m) vv[m] = v[base + (size_t)m * 256 + d];
#pragma unroll
    for (int n = 0; n < 8; ++n) {
      float s = 0.f;
#pragma unroll
      for (int m = 0; m < 8; ++m) s = fmaf(att[n][m], vv[m], s);
      o[base + (size_t)n * 256 + d] = s;
    }
  }
}

// out = layernorm(h + oproj) over last dim (256). One wave per row.
__global__ __launch_bounds__(256)
void add_ln_kernel(const float* __restrict__ h, const float* __restrict__ op,
                   const float* __restrict__ g, const float* __restrict__ be,
                   float* __restrict__ out, int rows)
{
  int row  = blockIdx.x * 8 + (threadIdx.x >> 5);
  int lane = threadIdx.x & 31;
  if (row >= rows) return;
  const float* __restrict__ hp = h  + (size_t)row * 256;
  const float* __restrict__ pp = op + (size_t)row * 256;
  float x[8]; float s = 0.f;
#pragma unroll
  for (int j = 0; j < 8; ++j) { x[j] = hp[lane + 32 * j] + pp[lane + 32 * j]; s += x[j]; }
  s = wave_sum(s);
  float mu = s * (1.f / 256.f);
  float s2 = 0.f;
#pragma unroll
  for (int j = 0; j < 8; ++j) { float d = x[j] - mu; s2 += d * d; }
  s2 = wave_sum(s2);
  float is = rsqrtf(s2 * (1.f / 256.f) + 1e-5f);
#pragma unroll
  for (int j = 0; j < 8; ++j) {
    int c = lane + 32 * j;
    out[(size_t)row * 256 + c] = (x[j] - mu) * is * g[c] + be[c];
  }
}

// Contact inhibition + final h + pooled mean. One block (8 waves) per batch.
__global__ __launch_bounds__(256)
void inhibit_kernel(const float* __restrict__ hln, const float* __restrict__ rho_p,
                    const float* __restrict__ eth_p, float* __restrict__ hout,
                    float* __restrict__ pooled, int B)
{
  int b    = blockIdx.x;
  int tid  = threadIdx.x;
  int node = tid >> 5;
  int lane = tid & 31;
  __shared__ float sh[2048];
  __shared__ float norms[8], ents[8], pen[8];

  const float* __restrict__ src = hln + (size_t)b * 2048;
  for (int i = tid; i < 2048; i += 256) sh[i] = src[i];
  __syncthreads();

  { // per-node norm + softmax entropy (wave `node` handles node `node`)
    const float* rw = sh + node * 256;
    float s2 = 0.f, mx = -3.402823e38f;
#pragma unroll
    for (int j = 0; j < 8; ++j) {
      float v = rw[lane + 32 * j];
      s2 += v * v; mx = fmaxf(mx, v);
    }
    s2 = wave_sum(s2); mx = wave_max(mx);
    float se = 0.f;
#pragma unroll
    for (int j = 0; j < 8; ++j) se += __expf(rw[lane + 32 * j] - mx);
    se = wave_sum(se);
    float lse = __logf(se);
    float en = 0.f;
#pragma unroll
    for (int j = 0; j < 8; ++j) {
      float lp = rw[lane + 32 * j] - mx - lse;
      en += -lp * __expf(lp);
    }
    en = wave_sum(en);
    if (lane == 0) { norms[node] = fmaxf(sqrtf(s2), 1e-12f); ents[node] = en; }
  }
  __syncthreads();

  { // pairwise cosine sim + entropy gap -> penalty
    float rho = rho_p[0], eth = eth_p[0];
    const float* rn = sh + node * 256;
    float cnt = 0.f;
    for (int m = 0; m < 8; ++m) {
      if (m == node) continue;
      const float* rm = sh + m * 256;
      float d = 0.f;
#pragma unroll
      for (int j = 0; j < 8; ++j) d = fmaf(rn[lane + 32 * j], rm[lane + 32 * j], d);
      d = wave_sum(d);
      float sim = d / (norms[node] * norms[m]);
      float es  = fabsf(ents[node] - ents[m]);
      if (sim > rho && es < eth) cnt += 1.f;
    }
    if (lane == 0) pen[node] = fminf(cnt, 1.f);
  }
  __syncthreads();

  for (int i = tid; i < 2048; i += 256) {
    int n = i >> 8;
    float v = sh[i] * (1.f - 0.5f * pen[n]);
    sh[i] = v;
    hout[(size_t)b * 2048 + i] = v;
  }
  __syncthreads();
  if (tid < 256) {
    float s = 0.f;
#pragma unroll
    for (int n = 0; n < 8; ++n) s += sh[n * 256 + tid];
    pooled[(size_t)b * 256 + tid] = s * 0.125f;
  }
}

// ---------------------------------------------------------------------------
extern "C" void kernel_launch(void* const* d_in, const int* in_sizes, int n_in,
                              void* d_out, int out_size, void* d_ws, size_t ws_size,
                              hipStream_t stream) {
  const float* x       = (const float*)d_in[0];
  const float* conv1_w = (const float*)d_in[1];
  const float* conv1_b = (const float*)d_in[2];
  const float* bn1_g   = (const float*)d_in[3];
  const float* bn1_b   = (const float*)d_in[4];
  const float* conv2_w = (const float*)d_in[5];
  const float* conv2_b = (const float*)d_in[6];
  const float* bn2_g   = (const float*)d_in[7];
  const float* bn2_b   = (const float*)d_in[8];
  const float* conv3_w = (const float*)d_in[9];
  const float* conv3_b = (const float*)d_in[10];
  const float* bn3_g   = (const float*)d_in[11];
  const float* bn3_b   = (const float*)d_in[12];
  const float* stem_w  = (const float*)d_in[13];
  const float* stem_b  = (const float*)d_in[14];
  const float* gates   = (const float*)d_in[15];
  const float* proj_w  = (const float*)d_in[16];
  const float* proj_b  = (const float*)d_in[17];
  const float* morph_w = (const float*)d_in[18];
  const float* morph_b = (const float*)d_in[19];
  const float* wq = (const float*)d_in[20]; const float* bq = (const float*)d_in[21];
  const float* wk = (const float*)d_in[22]; const float* bk = (const float*)d_in[23];
  const float* wv = (const float*)d_in[24]; const float* bv = (const float*)d_in[25];
  const float* wo = (const float*)d_in[26]; const float* bo = (const float*)d_in[27];
  const float* ln_g = (const float*)d_in[28]; const float* ln_b = (const float*)d_in[29];
  const float* head_w = (const float*)d_in[30]; const float* head_b = (const float*)d_in[31];
  const float* rho = (const float*)d_in[32]; const float* eth = (const float*)d_in[33];

  float* out = (float*)d_out;
  const int B = 512;

  // ----- workspace carve-out (floats) -----
  float* ws = (float*)d_ws;
  size_t off = 0;
  auto alloc = [&](size_t n) { float* p = ws + off; off += n; return p; };
  float* y1     = alloc((size_t)B * 32 * 32 * 32);
  float* y1p    = alloc((size_t)B * 32 * 16 * 16);
  float* y2     = alloc((size_t)B * 64 * 16 * 16);
  float* y2p    = alloc((size_t)B * 64 * 8 * 8);
  float* y3     = alloc((size_t)B * 128 * 8 * 8);
  float* z      = alloc((size_t)B * 2048);
  float* stems  = alloc((size_t)B * 1024);
  float* hvec   = alloc((size_t)B * 256);
  float* hnodes = alloc((size_t)B * 2048);
  float* qb     = alloc((size_t)B * 2048);
  float* kb     = alloc((size_t)B * 2048);
  float* vb     = alloc((size_t)B * 2048);
  float* ob     = alloc((size_t)B * 2048);
  float* oproj  = alloc((size_t)B * 2048);
  float* hln    = alloc((size_t)B * 2048);
  float* pooled = alloc((size_t)B * 256);
  float* stats  = alloc(256);

  // d_out layout: logits [B,100] | h [B,8,256] | morph [B,8,8]
  float* out_logits = out;
  float* out_h      = out + (size_t)B * 100;
  float* out_morph  = out + (size_t)B * 100 + (size_t)B * 2048;

  // ----- conv stack -----
  {
    int total = B * 32 * 32 * 32;
    conv3x3_same<<<(total + 255) / 256, 256, 0, stream>>>(x, conv1_w, conv1_b, y1, B, 3, 32, 32, 32);
    bn_stats<<<32, 256, 0, stream>>>(y1, stats, B, 32, 32 * 32);
    int tp = B * 32 * 16 * 16;
    bn_relu_pool<<<(tp + 255) / 256, 256, 0, stream>>>(y1, stats, bn1_g, bn1_b, y1p, B, 32, 32, 32, 0);
  }
  {
    int total = B * 64 * 16 * 16;
    conv3x3_same<<<(total + 255) / 256, 256, 0, stream>>>(y1p, conv2_w, conv2_b, y2, B, 32, 64, 16, 16);
    bn_stats<<<64, 256, 0, stream>>>(y2, stats, B, 64, 16 * 16);
    int tp = B * 64 * 8 * 8;
    bn_relu_pool<<<(tp + 255) / 256, 256, 0, stream>>>(y2, stats, bn2_g, bn2_b, y2p, B, 64, 16, 16, 0);
  }
  {
    int total = B * 128 * 8 * 8;
    conv3x3_same<<<(total + 255) / 256, 256, 0, stream>>>(y2p, conv3_w, conv3_b, y3, B, 64, 128, 8, 8);
    bn_stats<<<128, 256, 0, stream>>>(y3, stats, B, 128, 8 * 8);
    int tp = B * 128 * 4 * 4;
    bn_relu_pool<<<(tp + 255) / 256, 256, 0, stream>>>(y3, stats, bn3_g, bn3_b, z, B, 128, 8, 8, 1);
  }

  // ----- stem pool: relu(z @ stem_w^T + stem_b), then gate-mix -----
  gemm_wmma_nt<<<dim3(1024 / 64, B / 128), 256, 0, stream>>>(z, stem_w, stem_b, stems, B, 1024, 2048, 1);
  stems_combine<<<(B * 256 + 255) / 256, 256, 0, stream>>>(stems, gates, hvec, B);

  // ----- project to nodes: [B,256] @ proj_w^T -> [B,2048] -----
  gemm_wmma_nt<<<dim3(2048 / 64, B / 128), 256, 0, stream>>>(hvec, proj_w, proj_b, hnodes, B, 2048, 256, 0);

  // ----- morphogen stats + in-place update; morph -> d_out -----
  morph_kernel<<<B, 256, 0, stream>>>(hnodes, morph_w, morph_b, out_morph, B);

  // ----- attention: Q/K/V projections ([B*8,256] @ W^T) -----
  gemm_wmma_nt<<<dim3(256 / 64, (B * 8) / 128), 256, 0, stream>>>(hnodes, wq, bq, qb, B * 8, 256, 256, 0);
  gemm_wmma_nt<<<dim3(256 / 64, (B * 8) / 128), 256, 0, stream>>>(hnodes, wk, bk, kb, B * 8, 256, 256, 0);
  gemm_wmma_nt<<<dim3(256 / 64, (B * 8) / 128), 256, 0, stream>>>(hnodes, wv, bv, vb, B * 8, 256, 256, 0);
  attn_kernel<<<B * 4, 64, 0, stream>>>(qb, kb, vb, ob, B);
  gemm_wmma_nt<<<dim3(256 / 64, (B * 8) / 128), 256, 0, stream>>>(ob, wo, bo, oproj, B * 8, 256, 256, 0);

  // ----- residual + layernorm -----
  add_ln_kernel<<<(B * 8) / 8, 256, 0, stream>>>(hnodes, oproj, ln_g, ln_b, hln, B * 8);

  // ----- contact inhibition -> final h (d_out) + pooled -----
  inhibit_kernel<<<B, 256, 0, stream>>>(hln, rho, eth, out_h, pooled, B);

  // ----- head: pooled @ head_w^T + head_b -> logits (d_out) -----
  gemm_wmma_nt<<<dim3((100 + 63) / 64, B / 128), 256, 0, stream>>>(pooled, head_w, head_b, out_logits, B, 100, 256, 0);
}
